// MambaSS2D_28132035788850
// MI455X (gfx1250) — compile-verified, matrix operand order checked by reading, not run on hardware
//
#include <hip/hip_runtime.h>

#define DEV __device__ __forceinline__

typedef __bf16 bf16_t;
typedef __attribute__((ext_vector_type(16))) __bf16 v16bf;
typedef __attribute__((ext_vector_type(8)))  float  v8f;

constexpr int BATCH = 2;
constexpr int CH    = 64;     // d_model
constexpr int HH    = 64;
constexpr int WW    = 64;
constexpr int LL    = HH * WW; // 4096
constexpr int DI    = 128;    // d_inner
constexpr int DS    = 16;     // d_state
constexpr int LSTR  = 40;     // LDS staging row stride (ushorts); 80B = 16B-aligned rows

DEV unsigned short f2bf(float f) {
  unsigned int u = __float_as_uint(f);
  u += 0x7FFFu + ((u >> 16) & 1u);      // round-to-nearest-even
  return (unsigned short)(u >> 16);
}
DEV bf16_t us2bf(unsigned short u) {
  union { unsigned short s; bf16_t b; } x; x.s = u; return x.b;
}
DEV unsigned lds_off(const void* p) {   // low 32 bits of flat LDS address = LDS offset
  return (unsigned)(unsigned long long)(uintptr_t)p;
}

// ---------------------------------------------------------------------------
// Weight fp32 -> bf16 (with optional zero row-padding): dst is (nmat, padRows, cols)
// ---------------------------------------------------------------------------
__global__ __launch_bounds__(256)
void convert_pad_kernel(const float* __restrict__ src, unsigned short* __restrict__ dst,
                        int rows, int cols, int padRows, int nmat) {
  int idx = blockIdx.x * 256 + threadIdx.x;
  int total = nmat * padRows * cols;
  if (idx >= total) return;
  int m = idx / (padRows * cols);
  int r = (idx / cols) % padRows;
  int c = idx % cols;
  float v = (r < rows) ? src[((long)m * rows + r) * cols + c] : 0.0f;
  dst[idx] = f2bf(v);
}

// ---------------------------------------------------------------------------
// LayerNorm over channels: x (B,C,H,W) -> xn (B,L,C)
// ---------------------------------------------------------------------------
__global__ __launch_bounds__(64)
void ln_kernel(const float* __restrict__ x, const float* __restrict__ g,
               const float* __restrict__ bt, float* __restrict__ xn) {
  int blk = blockIdx.x;               // b*LL + l
  int b = blk / LL, l = blk % LL;
  int c = threadIdx.x;
  float v = x[((long)(b * CH + c)) * LL + l];
  __shared__ float s1[64], s2[64];
  s1[c] = v; s2[c] = v * v;
  __syncthreads();
  for (int off = 32; off > 0; off >>= 1) {
    if (c < off) { s1[c] += s1[c + off]; s2[c] += s2[c + off]; }
    __syncthreads();
  }
  float mean = s1[0] * (1.0f / CH);
  float var  = s2[0] * (1.0f / CH) - mean * mean;
  float r = rsqrtf(var + 1e-6f);
  xn[((long)b * LL + l) * CH + c] = (v - mean) * r * g[c] + bt[c];
}

// ---------------------------------------------------------------------------
// 3x3 depthwise conv -> bf16, row-major layout and transposed (col-major) layout
// ---------------------------------------------------------------------------
__global__ __launch_bounds__(64)
void dwconv_kernel(const float* __restrict__ xn, const float* __restrict__ w,
                   const float* __restrict__ bias,
                   unsigned short* __restrict__ xd_rm, unsigned short* __restrict__ xd_tr) {
  int blk = blockIdx.x;
  int b = blk / LL, l = blk % LL;
  int h = l >> 6, ww = l & 63;
  int c = threadIdx.x;
  float acc = bias[c];
#pragma unroll
  for (int i = -1; i <= 1; i++)
#pragma unroll
    for (int j = -1; j <= 1; j++) {
      int hh = h + i, w2 = ww + j;
      if (hh >= 0 && hh < HH && w2 >= 0 && w2 < WW)
        acc += xn[((long)b * LL + hh * WW + w2) * CH + c] * w[c * 9 + (i + 1) * 3 + (j + 1)];
    }
  unsigned short v = f2bf(acc);
  xd_rm[((long)b * LL + l) * CH + c] = v;
  xd_tr[((long)b * LL + (ww * HH + h)) * CH + c] = v;
}

// ---------------------------------------------------------------------------
// Generic bf16 WMMA GEMM: C[M,N] = A[M,K] * W[N,K]^T   (block tile 64x64)
// k-chunks of 32 double-buffered in LDS via GLOBAL_LOAD_ASYNC_TO_LDS_B128
// (ASYNCcnt-tracked direct memory->LDS copies, no VGPR round trip).
// MODE 0: fp32 store (ldC = N, column guard). MODE 1: bf16 store into the
// 4-direction concat buffer with spatial remap for transposed directions.
// ---------------------------------------------------------------------------
template<int MODE>
__global__ __launch_bounds__(256)
void gemm_bf16_kernel(const unsigned short* __restrict__ A, long sAb, long sAd,
                      const unsigned short* __restrict__ Wt, long sWd, int dirBase, int ndir,
                      float* __restrict__ Cf, long sCb, long sCd,
                      unsigned short* __restrict__ Cb,
                      int N, int K) {
  __shared__ unsigned short As[2][64 * LSTR];
  __shared__ unsigned short Bs[2][64 * LSTR];
  const int bz  = blockIdx.z;
  const int b   = bz / ndir;
  const int dl  = bz % ndir;
  const int dir = dirBase + dl;
  const unsigned short* Ab = A + (long)b * sAb + (long)dl * sAd;
  const unsigned short* Wb = Wt + (long)dir * sWd;
  const long cOff = (long)b * sCb + (long)dir * sCd;
  const int m0 = blockIdx.x * 64;
  const int n0 = blockIdx.y * 64;

  const int tid  = threadIdx.x;
  const int lane = tid & 31;
  const int wave = tid >> 5;
  const int mt   = wave >> 1;          // 0..3  (M sub-tile)
  const int ntb  = (wave & 1) * 2;     // 0 or 2 (N sub-tile base, 2 tiles per wave)
  const int hl   = lane >> 4;          // half (0/1)
  const int rl   = lane & 15;

  v8f acc0 = {}; v8f acc1 = {};

  const int lr = tid >> 2;             // staging row 0..63
  const int lc = (tid & 3) * 8;        // staging col base {0,8,16,24} (ushorts)

  const unsigned short* srcA = Ab + (long)(m0 + lr) * K + lc;
  const unsigned short* srcW = Wb + (long)(n0 + lr) * K + lc;
  const unsigned dstA0 = lds_off(&As[0][lr * LSTR + lc]);
  const unsigned dstW0 = lds_off(&Bs[0][lr * LSTR + lc]);
  const unsigned bufSz = (unsigned)(64 * LSTR * 2);  // bytes between buffers

  // issue one 16B async memory->LDS copy per thread for A and W of chunk kb
  auto issue = [&](int buf, int kb) {
    const unsigned short* pa = srcA + kb;
    const unsigned short* pw = srcW + kb;
    unsigned da = dstA0 + (unsigned)buf * bufSz;
    unsigned dw = dstW0 + (unsigned)buf * bufSz;
    asm volatile("global_load_async_to_lds_b128 %0, %1, off"
                 :: "v"(da), "v"(pa) : "memory");
    asm volatile("global_load_async_to_lds_b128 %0, %1, off"
                 :: "v"(dw), "v"(pw) : "memory");
    if (kb + 32 < K) {                 // global_prefetch_b8 for the chunk after next
      __builtin_prefetch(pa + 32, 0, 1);
      __builtin_prefetch(pw + 32, 0, 1);
    }
  };

  const int nChunks = K >> 5;
  issue(0, 0);                         // prologue: chunk 0 -> buffer 0
  for (int i = 0; i < nChunks; ++i) {
    const int buf = i & 1;
    const bool more = (i + 1 < nChunks);
    if (more) issue(buf ^ 1, (i + 1) << 5);   // overlap next chunk with compute
    // async loads complete in order: <=2 outstanding => this chunk's pair landed
    if (more) asm volatile("s_wait_asynccnt 0x2" ::: "memory");
    else      asm volatile("s_wait_asynccnt 0x0" ::: "memory");
    __syncthreads();

    v16bf af, bf0, bf1;
    const unsigned short* ar  = &As[buf][(mt * 16 + rl) * LSTR];
    const unsigned short* br0 = &Bs[buf][(ntb * 16 + rl) * LSTR];
    const unsigned short* br1 = &Bs[buf][((ntb + 1) * 16 + rl) * LSTR];
#pragma unroll
    for (int p = 0; p < 8; ++p) {
      // ISA 7.12.2: 16-bit A 16x32 layout; element pairs (2p,2p+1) -> k, k+1
      int k = (p < 4) ? (hl * 8 + 2 * p) : (16 + hl * 8 + 2 * (p - 4));
      unsigned int wa = *(const unsigned int*)(ar  + k);
      unsigned int w0 = *(const unsigned int*)(br0 + k);
      unsigned int w1 = *(const unsigned int*)(br1 + k);
      af[2 * p]      = us2bf((unsigned short)wa);
      af[2 * p + 1]  = us2bf((unsigned short)(wa >> 16));
      bf0[2 * p]     = us2bf((unsigned short)w0);
      bf0[2 * p + 1] = us2bf((unsigned short)(w0 >> 16));
      bf1[2 * p]     = us2bf((unsigned short)w1);
      bf1[2 * p + 1] = us2bf((unsigned short)(w1 >> 16));
    }
    acc0 = __builtin_amdgcn_wmma_f32_16x16x32_bf16(false, af, false, bf0, (short)0, acc0, false, false);
    acc1 = __builtin_amdgcn_wmma_f32_16x16x32_bf16(false, af, false, bf1, (short)0, acc1, false, false);
    __syncthreads();   // all waves done reading buf before it is refilled
  }

#pragma unroll
  for (int r = 0; r < 8; ++r) {
    int row = m0 + mt * 16 + hl * 8 + r;       // C/D layout: M = r + 8*half, N = lane&15
    int c0  = n0 + ntb * 16 + rl;
    int c1  = n0 + (ntb + 1) * 16 + rl;
    if (MODE == 0) {
      if (c0 < N) Cf[cOff + (long)row * N + c0] = acc0[r];
      if (c1 < N) Cf[cOff + (long)row * N + c1] = acc1[r];
    } else {
      // row == direction-layout position p; remap transposed dirs back to row-major l
      int l = (dir < 2) ? row : (((row & 63) << 6) + (row >> 6));
      long base = ((long)b * LL + l) * 256 + dir * 64;
      Cb[base + c0] = f2bf(acc0[r]);
      Cb[base + c1] = f2bf(acc1[r]);
    }
  }
}

// ---------------------------------------------------------------------------
// Causal depthwise conv1d (width 4) + SiLU. Reversal folded into tap indexing.
// ---------------------------------------------------------------------------
__global__ __launch_bounds__(128)
void conv1d_kernel(const float* __restrict__ xz, const float* __restrict__ cw,
                   const float* __restrict__ cb,
                   float* __restrict__ xcf, unsigned short* __restrict__ xcb) {
  int blk = blockIdx.x;                 // (b*4+dir)*LL + l
  int l = blk % LL;
  int bd = blk / LL;
  int dir = bd & 3;
  int d = threadIdx.x;
  bool rev = dir & 1;
  const float* base = xz + (long)bd * LL * 256;
  float acc = cb[dir * DI + d];
#pragma unroll
  for (int j = 0; j < 4; j++) {
    int pos = rev ? (l + 3 - j) : (l - 3 + j);
    if (pos >= 0 && pos < LL)
      acc += cw[(dir * DI + d) * 4 + j] * base[(long)pos * 256 + d];
  }
  float s = acc / (1.0f + __expf(-acc));
  long o = ((long)bd * LL + l) * DI + d;
  xcf[o] = s;
  xcb[o] = f2bf(s);
}

// ---------------------------------------------------------------------------
// dt = softplus(dt_raw @ dt_w^T + dt_b)   (K=4 -> scalar VALU)
// ---------------------------------------------------------------------------
__global__ __launch_bounds__(128)
void dt_kernel(const float* __restrict__ dbl, const float* __restrict__ dtw,
               const float* __restrict__ dtb, float* __restrict__ dt) {
  int blk = blockIdx.x;
  int l = blk % LL;
  int bd = blk / LL;
  int dir = bd & 3;
  int d = threadIdx.x;
  const float* row = dbl + ((long)bd * LL + l) * 48;
  float v = dtb[dir * DI + d];
#pragma unroll
  for (int r = 0; r < 4; r++) v += row[r] * dtw[(dir * DI + d) * 4 + r];
  float sp = (v > 20.0f) ? v : log1pf(__expf(v));
  dt[((long)bd * LL + l) * DI + d] = sp;
}

// ---------------------------------------------------------------------------
// Selective scan: one lane per (channel d, state s); 16-lane shuffle reduction
// per step; B/C/dt/xc chunk-staged in LDS; fused (y + D*xc)*silu(z) -> bf16.
// Grid: 32 blocks = B * 4 dirs * 4 channel-splits, block = 512 (32 d x 16 s).
// ---------------------------------------------------------------------------
__global__ __launch_bounds__(512)
void scan_kernel(const float* __restrict__ dbl, const float* __restrict__ dtb,
                 const float* __restrict__ xcf, const float* __restrict__ xz,
                 const float* __restrict__ A_log, const float* __restrict__ D_ssm,
                 unsigned short* __restrict__ ymod) {
  int bz  = blockIdx.x;
  int dsp = bz & 3;
  int dir = (bz >> 2) & 3;
  int b   = bz >> 4;
  int bd  = b * 4 + dir;
  int tid = threadIdx.x;
  int grp = tid >> 4;                  // local channel 0..31
  int s   = tid & 15;                  // state
  int d   = dsp * 32 + grp;
  float Aval = -__expf(A_log[((long)dir * DI + d) * DS + s]);
  float Dval = D_ssm[dir * DI + d];
  float h = 0.0f;
  bool rev = dir & 1;
  long baseRow = (long)bd * LL;

  __shared__ float sBC[64 * 32];       // per step: 16 B then 16 C
  __shared__ float sDT[64 * 32];       // [step][local d]
  __shared__ float sXC[64 * 32];

  for (int t0 = 0; t0 < LL; t0 += 64) {
    for (int f = tid; f < 64 * 32; f += 512) {
      int st = f >> 5, q = f & 31;
      int tt = t0 + st;
      int l = rev ? (LL - 1 - tt) : tt;
      sBC[f] = dbl[(baseRow + l) * 48 + 4 + q];
      long ro = (baseRow + l) * DI + dsp * 32 + q;
      sDT[f] = dtb[ro];
      sXC[f] = xcf[ro];
    }
    __syncthreads();
    for (int st = 0; st < 64; ++st) {
      float dtv = sDT[st * 32 + grp];
      float xcv = sXC[st * 32 + grp];
      float Bs  = sBC[st * 32 + s];
      float Cs  = sBC[st * 32 + 16 + s];
      h = __expf(dtv * Aval) * h + (dtv * xcv) * Bs;
      float part = h * Cs;
      part += __shfl_xor(part, 1, 16);
      part += __shfl_xor(part, 2, 16);
      part += __shfl_xor(part, 4, 16);
      part += __shfl_xor(part, 8, 16);
      if (s == 0) {
        int tt = t0 + st;
        int l = rev ? (LL - 1 - tt) : tt;
        float zv = xz[(baseRow + l) * 256 + 128 + d];
        float yv = (part + Dval * xcv) * (zv / (1.0f + __expf(-zv)));
        ymod[(baseRow + l) * DI + d] = f2bf(yv);
      }
    }
    __syncthreads();
  }
}

// ---------------------------------------------------------------------------
// Channel attention: mean over L, 64->16 GELU, 16->64 sigmoid
// ---------------------------------------------------------------------------
__global__ __launch_bounds__(256)
void ca_kernel(const float* __restrict__ yall, const float* __restrict__ ob,
               const float* __restrict__ w1, const float* __restrict__ b1,
               const float* __restrict__ w2, const float* __restrict__ b2,
               float* __restrict__ g) {
  int b = blockIdx.x;
  int tid = threadIdx.x;
  int c = tid & 63, part = tid >> 6;
  float ssum = 0.0f;
  for (int l = part; l < LL; l += 4)
    ssum += yall[((long)b * LL + l) * CH + c];
  __shared__ float red[256];
  __shared__ float mc[64];
  __shared__ float hid[16];
  red[tid] = ssum;
  __syncthreads();
  if (part == 0)
    mc[c] = (red[c] + red[64 + c] + red[128 + c] + red[192 + c]) * (1.0f / LL) + ob[c];
  __syncthreads();
  if (tid < 16) {
    float a = b1[tid];
    for (int i = 0; i < 64; i++) a += mc[i] * w1[tid * 64 + i];
    hid[tid] = 0.5f * a * (1.0f + erff(a * 0.70710678118f));   // exact GELU
  }
  __syncthreads();
  if (tid < 64) {
    float a = b2[tid];
    for (int j = 0; j < 16; j++) a += hid[j] * w2[tid * 16 + j];
    g[b * CH + tid] = 1.0f / (1.0f + __expf(-a));
  }
}

// ---------------------------------------------------------------------------
// Residual add with channel-attention gate: out (B,C,H,W)
// ---------------------------------------------------------------------------
__global__ __launch_bounds__(256)
void out_kernel(const float* __restrict__ x, const float* __restrict__ yall,
                const float* __restrict__ ob, const float* __restrict__ g,
                float* __restrict__ out) {
  long idx = (long)blockIdx.x * 256 + threadIdx.x;
  long total = (long)BATCH * CH * LL;
  if (idx >= total) return;
  int l = idx % LL;
  int c = (idx / LL) % CH;
  int b = idx / ((long)LL * CH);
  float y = yall[((long)b * LL + l) * CH + c] + ob[c];
  out[idx] = x[idx] + y * g[b * CH + c];
}

// ---------------------------------------------------------------------------
extern "C" void kernel_launch(void* const* d_in, const int* in_sizes, int n_in,
                              void* d_out, int out_size, void* d_ws, size_t ws_size,
                              hipStream_t stream) {
  const float* x      = (const float*)d_in[0];
  const float* ln_g   = (const float*)d_in[1];
  const float* ln_b   = (const float*)d_in[2];
  const float* dw_w   = (const float*)d_in[3];
  const float* dw_b   = (const float*)d_in[4];
  const float* in_w   = (const float*)d_in[5];
  const float* conv_w = (const float*)d_in[6];
  const float* conv_b = (const float*)d_in[7];
  const float* xproj_w= (const float*)d_in[8];
  const float* dt_w   = (const float*)d_in[9];
  const float* dt_b   = (const float*)d_in[10];
  const float* A_log  = (const float*)d_in[11];
  const float* D_ssm  = (const float*)d_in[12];
  const float* ssm_w  = (const float*)d_in[13];
  const float* outp_w = (const float*)d_in[14];
  const float* outp_b = (const float*)d_in[15];
  const float* ca_w1  = (const float*)d_in[16];
  const float* ca_b1  = (const float*)d_in[17];
  const float* ca_w2  = (const float*)d_in[18];
  const float* ca_b2  = (const float*)d_in[19];
  float* out = (float*)d_out;

  char* ws = (char*)d_ws;
  size_t off = 0;
  auto alloc = [&](size_t bytes) -> char* {
    char* p = ws + off;
    off += (bytes + 255) & ~(size_t)255;
    return p;
  };
  float*          xn     = (float*)         alloc((size_t)BATCH * LL * CH * 4);
  unsigned short* xd_rm  = (unsigned short*)alloc((size_t)BATCH * LL * CH * 2);
  unsigned short* xd_tr  = (unsigned short*)alloc((size_t)BATCH * LL * CH * 2);
  unsigned short* w_in   = (unsigned short*)alloc((size_t)4 * 256 * 64 * 2);
  unsigned short* w_xp   = (unsigned short*)alloc((size_t)4 * 64 * 128 * 2);
  unsigned short* w_out  = (unsigned short*)alloc((size_t)4 * 64 * 128 * 2);
  unsigned short* w_fin  = (unsigned short*)alloc((size_t)64 * 256 * 2);
  float*          xz     = (float*)         alloc((size_t)BATCH * 4 * LL * 256 * 4);
  float*          xc_f   = (float*)         alloc((size_t)BATCH * 4 * LL * DI * 4);
  unsigned short* xc_bf  = (unsigned short*)alloc((size_t)BATCH * 4 * LL * DI * 2);
  float*          dbl    = (float*)         alloc((size_t)BATCH * 4 * LL * 48 * 4);
  float*          dtbuf  = (float*)         alloc((size_t)BATCH * 4 * LL * DI * 4);
  unsigned short* ymod   = (unsigned short*)alloc((size_t)BATCH * 4 * LL * DI * 2);
  unsigned short* dircat = (unsigned short*)alloc((size_t)BATCH * LL * 256 * 2);
  float*          yall   = (float*)         alloc((size_t)BATCH * LL * CH * 4);
  float*          gbuf   = (float*)         alloc((size_t)BATCH * CH * 4);

  // weight conversion (bf16, padded)
  convert_pad_kernel<<<(4 * 256 * 64 + 255) / 256, 256, 0, stream>>>(in_w,    w_in,  256, 64, 256, 4);
  convert_pad_kernel<<<(4 * 64 * 128 + 255) / 256, 256, 0, stream>>>(xproj_w, w_xp,  36, 128, 64, 4);
  convert_pad_kernel<<<(4 * 64 * 128 + 255) / 256, 256, 0, stream>>>(ssm_w,   w_out, 64, 128, 64, 4);
  convert_pad_kernel<<<(1 * 64 * 256 + 255) / 256, 256, 0, stream>>>(outp_w,  w_fin, 64, 256, 64, 1);

  // LN + depthwise conv
  ln_kernel<<<BATCH * LL, 64, 0, stream>>>(x, ln_g, ln_b, xn);
  dwconv_kernel<<<BATCH * LL, 64, 0, stream>>>(xn, dw_w, dw_b, xd_rm, xd_tr);

  // in_proj: dirs 0,1 use row-major layout; dirs 2,3 use transposed layout
  gemm_bf16_kernel<0><<<dim3(64, 4, BATCH * 2), 256, 0, stream>>>(
      xd_rm, (long)LL * CH, 0,
      w_in, (long)256 * 64, 0, 2,
      xz, (long)4 * LL * 256, (long)LL * 256, nullptr, 256, 64);
  gemm_bf16_kernel<0><<<dim3(64, 4, BATCH * 2), 256, 0, stream>>>(
      xd_tr, (long)LL * CH, 0,
      w_in, (long)256 * 64, 2, 2,
      xz, (long)4 * LL * 256, (long)LL * 256, nullptr, 256, 64);

  // conv1d + silu
  conv1d_kernel<<<BATCH * 4 * LL, 128, 0, stream>>>(xz, conv_w, conv_b, xc_f, xc_bf);

  // x_proj (N=36 padded to 48/64)
  gemm_bf16_kernel<0><<<dim3(64, 1, BATCH * 4), 256, 0, stream>>>(
      xc_bf, (long)4 * LL * DI, (long)LL * DI,
      w_xp, (long)64 * 128, 0, 4,
      dbl, (long)4 * LL * 48, (long)LL * 48, nullptr, 48, 128);

  // dt projection + softplus
  dt_kernel<<<BATCH * 4 * LL, 128, 0, stream>>>(dbl, dt_w, dt_b, dtbuf);

  // selective scan (fused gating epilogue -> bf16)
  scan_kernel<<<BATCH * 4 * 4, 512, 0, stream>>>(dbl, dtbuf, xc_f, xz, A_log, D_ssm, ymod);

  // out_proj with remap into 4-direction concat buffer (bf16)
  gemm_bf16_kernel<1><<<dim3(64, 1, BATCH * 4), 256, 0, stream>>>(
      ymod, (long)4 * LL * DI, (long)LL * DI,
      w_out, (long)64 * 128, 0, 4,
      nullptr, 0, 0, dircat, 64, 128);

  // final 256 -> 64 combine
  gemm_bf16_kernel<0><<<dim3(64, 1, BATCH), 256, 0, stream>>>(
      dircat, (long)LL * 256, 0,
      w_fin, 0, 0, 1,
      yall, (long)LL * CH, 0, nullptr, 64, 256);

  // channel attention + residual
  ca_kernel<<<BATCH, 256, 0, stream>>>(yall, outp_b, ca_w1, ca_b1, ca_w2, ca_b2, gbuf);
  out_kernel<<<(BATCH * CH * LL + 255) / 256, 256, 0, stream>>>(x, yall, outp_b, gbuf, out);
}
